// ContinuousThoughtMachine_78056735637991
// MI455X (gfx1250) — matrix operand order repaired and checked by reading, not compiled, not run on hardware
//
#include <hip/hip_runtime.h>

// ---- model dims ----
#define Bb   64
#define Ss   196
#define Ee   512
#define Dd   2048
#define Mm   25
#define HNh  32
#define NHh  8
#define Tt   50
#define NSs  512
#define OUTn 1000
#define HDd  64

typedef __attribute__((ext_vector_type(16))) __bf16 v16bf;
typedef __attribute__((ext_vector_type(8)))  __bf16 v8bf;
typedef __attribute__((ext_vector_type(4)))  __bf16 v4bf;
typedef __attribute__((ext_vector_type(8)))  float  v8f;

__device__ __forceinline__ float sigmoidf_(float x) { return 1.0f / (1.0f + __expf(-x)); }

// ---------------------------------------------------------------------------
// WMMA GEMM: C(M,N) = A(M,K) @ W(K,N) + bias.
//  - A fp32 in global, staged per 32x32 tile into LDS as bf16 (double-buffered),
//    fragment = two contiguous 16B runs (2x ds_load_b128).
//  - W pre-swizzled bf16 in global: Wsw[kt][n][kk], kk contiguous => B fragment
//    is one contiguous 32B load (2x global_load_b128, L2-hot), register-
//    prefetched one K-step ahead.
//  - Block: 256 thr = 8 waves; block tile 32x128; each wave a 16x32 C tile
//    (one A fragment feeds two WMMAs). Branch-free inner loop, peeled epilogue.
// Requirements: M % 32 == 0, K % 32 == 0 and K >= 32,
// Npad % 128 == 0 (true for all call sites).
// ---------------------------------------------------------------------------
#define GBM 32
#define GBN 128
#define GBK 32
#define ASTRIDE 40   // bf16 elements per LDS row (80B, 16B-aligned rows)

__global__ __launch_bounds__(256)
void gemm_wmma_bf16(const float* __restrict__ A, int lda,
                    const __bf16* __restrict__ Wsw,   // swizzled (K/32, Npad, 32)
                    const float* __restrict__ bias,
                    float* __restrict__ C,
                    int N, int Npad, int K)
{
    __shared__ __bf16 As[2][GBM][ASTRIDE];

    const int tid  = threadIdx.x;
    const int lane = tid & 31;
    const int wid  = tid >> 5;
    const int rowbase = blockIdx.y * GBM;
    const int colbase = blockIdx.x * GBN;
    const int wm  = (wid >> 2) * 16;   // 0 or 16: row tile within block
    const int cg  = (wid & 3) * 32;    // 0,32,64,96: col group within block
    const int hl  = lane >> 4;
    const int l16 = lane & 15;

    // cooperative A stage: each thread owns one float4 (row lr, cols lk..lk+3)
    const int lr = tid >> 3;           // 0..31
    const int lk = (tid & 7) * 4;      // 0..28 step 4
    const float* arow_g = A + (size_t)(rowbase + lr) * lda + lk;

    // stage K-tile 0
    {
        float4 va = *(const float4*)(arow_g);
        v4bf t; t[0] = (__bf16)va.x; t[1] = (__bf16)va.y;
                t[2] = (__bf16)va.z; t[3] = (__bf16)va.w;
        *(v4bf*)&As[0][lr][lk] = t;
    }
    __syncthreads();

    const __bf16* wb0 = Wsw + (size_t)(colbase + cg + l16) * 32 + hl * 16;
    const __bf16* wb1 = wb0 + 16 * 32;                 // +16 columns
    const size_t  wkstep = (size_t)Npad * 32;          // elements per K-tile

    // register prefetch of B fragments for K-tile 0
    v16bf wf0 = *(const v16bf*)(wb0);
    v16bf wf1 = *(const v16bf*)(wb1);

    v8f acc0 = {}, acc1 = {};
    const __bf16* afrag_lo = &As[0][wm + l16][hl * 16];
    const size_t  bufstep  = (size_t)GBM * ASTRIDE;

    int buf = 0;
    for (int kb = 0; kb + GBK < K; kb += GBK) {
        // stage next A tile (overlaps WMMA)
        {
            float4 va = *(const float4*)(arow_g + kb + GBK);
            v4bf t; t[0] = (__bf16)va.x; t[1] = (__bf16)va.y;
                    t[2] = (__bf16)va.z; t[3] = (__bf16)va.w;
            *(v4bf*)&As[buf ^ 1][lr][lk] = t;
        }
        // prefetch next B fragments into registers
        const size_t knext = (size_t)((kb >> 5) + 1) * wkstep;
        v16bf nf0 = *(const v16bf*)(wb0 + knext);
        v16bf nf1 = *(const v16bf*)(wb1 + knext);

        // A fragment: two contiguous 16B LDS loads
        const __bf16* ap = afrag_lo + (size_t)buf * bufstep;
        v8bf alo = *(const v8bf*)(ap);
        v8bf ahi = *(const v8bf*)(ap + 8);
        v16bf af = __builtin_shufflevector(alo, ahi,
                       0, 1, 2, 3, 4, 5, 6, 7, 8, 9, 10, 11, 12, 13, 14, 15);

        acc0 = __builtin_amdgcn_wmma_f32_16x16x32_bf16(
                   false, af, false, wf0, (short)0, acc0, false, false);
        acc1 = __builtin_amdgcn_wmma_f32_16x16x32_bf16(
                   false, af, false, wf1, (short)0, acc1, false, false);
        __syncthreads();
        wf0 = nf0; wf1 = nf1;
        buf ^= 1;
    }
    // epilogue: last K-tile (fragments already staged/prefetched)
    {
        const __bf16* ap = afrag_lo + (size_t)buf * bufstep;
        v8bf alo = *(const v8bf*)(ap);
        v8bf ahi = *(const v8bf*)(ap + 8);
        v16bf af = __builtin_shufflevector(alo, ahi,
                       0, 1, 2, 3, 4, 5, 6, 7, 8, 9, 10, 11, 12, 13, 14, 15);
        acc0 = __builtin_amdgcn_wmma_f32_16x16x32_bf16(
                   false, af, false, wf0, (short)0, acc0, false, false);
        acc1 = __builtin_amdgcn_wmma_f32_16x16x32_bf16(
                   false, af, false, wf1, (short)0, acc1, false, false);
    }

    // C layout: VGPR r -> row = r + hl*8, col = lane%16 (per 16x16 tile)
    #pragma unroll
    for (int r = 0; r < 8; ++r) {
        int m  = rowbase + wm + hl * 8 + r;
        int n0 = colbase + cg + l16;
        int n1 = n0 + 16;
        if (n0 < N) C[(size_t)m * N + n0] = acc0[r] + bias[n0];
        if (n1 < N) C[(size_t)m * N + n1] = acc1[r] + bias[n1];
    }
}

// Pack fp32 weights (K x N) into swizzled bf16 (K/32, Npad, 32), zero-padded.
__global__ void pack_w(const float* __restrict__ W, __bf16* __restrict__ Wsw,
                       int N, int Npad)
{
    int idx = blockIdx.x * 256 + threadIdx.x;   // < K*Npad
    int kk = idx & 31;
    int n  = (idx >> 5) % Npad;
    int kt = idx / (32 * Npad);
    float v = (n < N) ? W[(size_t)(kt * 32 + kk) * N + n] : 0.0f;
    Wsw[idx] = (__bf16)v;
}

// ---------------------------------------------------------------------------
// Row-wise LayerNorm over 512 columns, in place. One block (256 thr) per row.
// ---------------------------------------------------------------------------
__global__ __launch_bounds__(256)
void ln_rows512(float* __restrict__ data, const float* __restrict__ g,
                const float* __restrict__ bta)
{
    __shared__ float r1[256], r2[256];
    const int tid = threadIdx.x;
    const size_t base = (size_t)blockIdx.x * Ee;
    float x0 = data[base + tid];
    float x1 = data[base + 256 + tid];
    r1[tid] = x0 + x1;
    r2[tid] = x0 * x0 + x1 * x1;
    __syncthreads();
    for (int off = 128; off; off >>= 1) {
        if (tid < off) { r1[tid] += r1[tid + off]; r2[tid] += r2[tid + off]; }
        __syncthreads();
    }
    float mean = r1[0] * (1.0f / Ee);
    float var  = r2[0] * (1.0f / Ee) - mean * mean;
    float rstd = rsqrtf(var + 1e-5f);
    data[base + tid]       = (x0 - mean) * rstd * g[tid]       + bta[tid];
    data[base + 256 + tid] = (x1 - mean) * rstd * g[256 + tid] + bta[256 + tid];
}

// ---------------------------------------------------------------------------
// Attention: one block per (b,h). scores over S=196, softmax, weighted V sum.
// ---------------------------------------------------------------------------
__global__ __launch_bounds__(256)
void attn_kernel(const float* __restrict__ qh, const float* __restrict__ kh,
                 const float* __restrict__ vh, float* __restrict__ attno)
{
    __shared__ float qs[HDd];
    __shared__ float sc[256];
    __shared__ float red[256];
    const int b = blockIdx.x / NHh, h = blockIdx.x % NHh;
    const int tid = threadIdx.x;

    if (tid < HDd) qs[tid] = qh[(size_t)b * Ee + h * HDd + tid];
    __syncthreads();

    float v = -3.0e38f;
    if (tid < Ss) {
        const float* kp = kh + (((size_t)b * Ss + tid) * NHh + h) * HDd;
        float a = 0.0f;
        #pragma unroll 8
        for (int d = 0; d < HDd; ++d) a += qs[d] * kp[d];
        v = a * 0.125f;                       // 1/sqrt(HD)
    }
    sc[tid]  = v;
    red[tid] = v;
    __syncthreads();
    for (int off = 128; off; off >>= 1) {
        if (tid < off) red[tid] = fmaxf(red[tid], red[tid + off]);
        __syncthreads();
    }
    float mx = red[0];
    __syncthreads();
    float e = 0.0f;
    if (tid < Ss) { e = __expf(sc[tid] - mx); sc[tid] = e; }
    red[tid] = e;
    __syncthreads();
    for (int off = 128; off; off >>= 1) {
        if (tid < off) red[tid] += red[tid + off];
        __syncthreads();
    }
    float inv = 1.0f / red[0];
    __syncthreads();
    if (tid < HDd) {
        const float* vp = vh + ((size_t)b * Ss * NHh + h) * HDd + tid;
        float a = 0.0f;
        for (int s2 = 0; s2 < Ss; ++s2) a += sc[s2] * vp[(size_t)s2 * Ee];
        attno[(size_t)b * Ee + h * HDd + tid] = a * inv;
    }
}

// ---------------------------------------------------------------------------
// Sync pair update (shared for action/out paths). idx over B*NS.
// ---------------------------------------------------------------------------
__global__ void sync_update(const float* __restrict__ act,
                            const int* __restrict__ il, const int* __restrict__ ir,
                            const float* __restrict__ r,
                            float* __restrict__ a, float* __restrict__ bm,
                            float* __restrict__ syncv, float* __restrict__ final_out)
{
    int idx = blockIdx.x * 256 + threadIdx.x;      // < B*NS
    int b = idx / NSs, i = idx % NSs;
    float rr = r[i];
    float p  = act[(size_t)b * Dd + il[i]] * act[(size_t)b * Dd + ir[i]];
    float av = rr * a[idx] + p;
    float bv = rr * bm[idx] + 1.0f;
    a[idx] = av; bm[idx] = bv;
    float sv = av * rsqrtf(bv);
    syncv[idx] = sv;
    if (final_out) final_out[idx] = sv;
}

// pre = concat(attnO (B,E), act (B,D)) -> (B, E+D)
__global__ void concat_pre(const float* __restrict__ attnO,
                           const float* __restrict__ act, float* __restrict__ pre)
{
    int idx = blockIdx.x * 256 + threadIdx.x;      // < B*(E+D)
    int b = idx / (Ee + Dd), k = idx % (Ee + Dd);
    pre[idx] = (k < Ee) ? attnO[(size_t)b * Ee + k]
                        : act[(size_t)b * Dd + (k - Ee)];
}

// ---------------------------------------------------------------------------
// GLU(4096->2048) + LayerNorm, write s into trace ring slot `pos`.
// ---------------------------------------------------------------------------
__global__ __launch_bounds__(256)
void glu_ln_trace(const float* __restrict__ praw, const float* __restrict__ g,
                  const float* __restrict__ bb, float* __restrict__ trace, int pos)
{
    __shared__ float r1[256], r2[256];
    const int b = blockIdx.x, tid = threadIdx.x;
    float gv[8];
    float s = 0.0f, q = 0.0f;
    #pragma unroll
    for (int i = 0; i < 8; ++i) {
        int d = tid + i * 256;
        float x = praw[(size_t)b * 2 * Dd + d];
        float y = praw[(size_t)b * 2 * Dd + Dd + d];
        float u = x * sigmoidf_(y);
        gv[i] = u; s += u; q += u * u;
    }
    r1[tid] = s; r2[tid] = q;
    __syncthreads();
    for (int off = 128; off; off >>= 1) {
        if (tid < off) { r1[tid] += r1[tid + off]; r2[tid] += r2[tid + off]; }
        __syncthreads();
    }
    float mean = r1[0] * (1.0f / Dd);
    float var  = r2[0] * (1.0f / Dd) - mean * mean;
    float rstd = rsqrtf(var + 1e-5f);
    #pragma unroll
    for (int i = 0; i < 8; ++i) {
        int d = tid + i * 256;
        float val = (gv[i] - mean) * rstd * g[d] + bb[d];
        trace[((size_t)b * Dd + d) * Mm + pos] = val;
    }
}

// ---------------------------------------------------------------------------
// Per-neuron MLP over trace ring history: 25 -> 64 -> GLU(32) -> 2 -> GLU(1).
// ---------------------------------------------------------------------------
__global__ __launch_bounds__(256)
void nlm_kernel(const float* __restrict__ trace,
                const float* __restrict__ w1, const float* __restrict__ b1,
                const float* __restrict__ w2, const float* __restrict__ b2,
                float* __restrict__ act, int start)
{
    int idx = blockIdx.x * 256 + threadIdx.x;      // < B*D
    int d = idx % Dd;
    float trv[Mm];
    {
        int p = start;
        size_t base = (size_t)idx * Mm;
        #pragma unroll
        for (int m = 0; m < Mm; ++m) {
            trv[m] = trace[base + p];
            p = (p + 1 == Mm) ? 0 : p + 1;
        }
    }
    float o0 = b2[d * 2 + 0], o1 = b2[d * 2 + 1];
    #pragma unroll 4
    for (int j = 0; j < HNh; ++j) {
        float aacc = b1[d * 64 + j];
        float bacc = b1[d * 64 + j + HNh];
        const float* w1a = w1 + (size_t)j * Dd + d;
        const float* w1b = w1 + (size_t)(j + HNh) * Dd + d;
        #pragma unroll
        for (int m = 0; m < Mm; ++m) {
            aacc += trv[m] * w1a[(size_t)m * 64 * Dd];
            bacc += trv[m] * w1b[(size_t)m * 64 * Dd];
        }
        float gl = aacc * sigmoidf_(bacc);
        o0 += gl * w2[(size_t)(j * 2 + 0) * Dd + d];
        o1 += gl * w2[(size_t)(j * 2 + 1) * Dd + d];
    }
    act[idx] = o0 * sigmoidf_(o1);
}

// ---------------------------------------------------------------------------
// Per-row: write raw predictions (strided by T), entropy-based certainty.
// ---------------------------------------------------------------------------
__global__ __launch_bounds__(256)
void pred_final(const float* __restrict__ pred, float* __restrict__ out_pred,
                float* __restrict__ out_cert, int t)
{
    __shared__ float red[256];
    const int b = blockIdx.x, tid = threadIdx.x;
    float pv[4];
    float mx = -3.0e38f;
    #pragma unroll
    for (int i = 0; i < 4; ++i) {
        int c = tid + i * 256;
        pv[i] = (c < OUTn) ? pred[(size_t)b * OUTn + c] : -3.0e38f;
        mx = fmaxf(mx, pv[i]);
    }
    red[tid] = mx; __syncthreads();
    for (int off = 128; off; off >>= 1) {
        if (tid < off) red[tid] = fmaxf(red[tid], red[tid + off]);
        __syncthreads();
    }
    mx = red[0]; __syncthreads();
    float se = 0.0f;
    #pragma unroll
    for (int i = 0; i < 4; ++i) {
        int c = tid + i * 256;
        if (c < OUTn) se += __expf(pv[i] - mx);
    }
    red[tid] = se; __syncthreads();
    for (int off = 128; off; off >>= 1) {
        if (tid < off) red[tid] += red[tid + off];
        __syncthreads();
    }
    float lse = mx + __logf(red[0]); __syncthreads();
    float acc = 0.0f;
    #pragma unroll
    for (int i = 0; i < 4; ++i) {
        int c = tid + i * 256;
        if (c < OUTn) { float lp = pv[i] - lse; acc += __expf(lp) * lp; }
    }
    red[tid] = acc; __syncthreads();
    for (int off = 128; off; off >>= 1) {
        if (tid < off) red[tid] += red[tid + off];
        __syncthreads();
    }
    float ne = -red[0] * (1.0f / 6.907755278982137f);   // / ln(1000)
    #pragma unroll
    for (int i = 0; i < 4; ++i) {
        int c = tid + i * 256;
        if (c < OUTn) out_pred[(size_t)b * OUTn * Tt + (size_t)c * Tt + t] = pv[i];
    }
    if (tid == 0) {
        out_cert[(size_t)b * 2 * Tt + t]      = ne;
        out_cert[(size_t)b * 2 * Tt + Tt + t] = 1.0f - ne;
    }
}

// ---- init kernels ----
__global__ void init_bd(const float* __restrict__ sa, const float* __restrict__ st,
                        float* __restrict__ act, float* __restrict__ trace)
{
    int idx = blockIdx.x * 256 + threadIdx.x;      // < B*D
    int d = idx % Dd;
    act[idx] = sa[d];
    #pragma unroll
    for (int m = 0; m < Mm; ++m)
        trace[(size_t)idx * Mm + m] = st[(size_t)d * Mm + m];
}

__global__ void init_ns(const float* __restrict__ sa,
                        const int* __restrict__ ol, const int* __restrict__ orr,
                        float* __restrict__ aA, float* __restrict__ bA,
                        float* __restrict__ aO, float* __restrict__ bO)
{
    int idx = blockIdx.x * 256 + threadIdx.x;      // < B*NS
    int i = idx % NSs;
    aA[idx] = 0.0f; bA[idx] = 0.0f;
    aO[idx] = sa[ol[i]] * sa[orr[i]];
    bO[idx] = 1.0f;
}

__global__ void init_decay(const float* __restrict__ da, const float* __restrict__ dro,
                           float* __restrict__ rA, float* __restrict__ rO)
{
    int i = blockIdx.x * 256 + threadIdx.x;
    if (i < NSs) {
        rA[i] = __expf(-fminf(fmaxf(da[i],  0.0f), 15.0f));
        rO[i] = __expf(-fminf(fmaxf(dro[i], 0.0f), 15.0f));
    }
}

// ---------------------------------------------------------------------------
extern "C" void kernel_launch(void* const* d_in, const int* in_sizes, int n_in,
                              void* d_out, int out_size, void* d_ws, size_t ws_size,
                              hipStream_t stream)
{
    (void)in_sizes; (void)n_in; (void)out_size; (void)ws_size;
    const float* x        = (const float*)d_in[0];
    const float* Wkv      = (const float*)d_in[1];
    const float* bkv      = (const float*)d_in[2];
    const float* g_kv     = (const float*)d_in[3];
    const float* b_kv     = (const float*)d_in[4];
    const float* Wq       = (const float*)d_in[5];
    const float* bq       = (const float*)d_in[6];
    const float* Wqa      = (const float*)d_in[7];
    const float* bqa      = (const float*)d_in[8];
    const float* Wka      = (const float*)d_in[9];
    const float* bka      = (const float*)d_in[10];
    const float* Wva      = (const float*)d_in[11];
    const float* bva      = (const float*)d_in[12];
    const float* Wo       = (const float*)d_in[13];
    const float* bo       = (const float*)d_in[14];
    const float* Wsyn     = (const float*)d_in[15];
    const float* bsyn     = (const float*)d_in[16];
    const float* g_syn    = (const float*)d_in[17];
    const float* b_syn    = (const float*)d_in[18];
    const float* w1       = (const float*)d_in[19];
    const float* b1       = (const float*)d_in[20];
    const float* w2       = (const float*)d_in[21];
    const float* b2       = (const float*)d_in[22];
    const float* start_act   = (const float*)d_in[23];
    const float* start_trace = (const float*)d_in[24];
    const float* decay_action= (const float*)d_in[25];
    const float* decay_out   = (const float*)d_in[26];
    const float* Wout     = (const float*)d_in[27];
    const float* bout     = (const float*)d_in[28];
    const int* idx_act_l  = (const int*)d_in[29];
    const int* idx_act_r  = (const int*)d_in[30];
    const int* idx_out_l  = (const int*)d_in[31];
    const int* idx_out_r  = (const int*)d_in[32];

    float* out      = (float*)d_out;
    float* out_pred = out;                                   // (B,OUT,T)
    float* out_cert = out + (size_t)Bb * OUTn * Tt;          // (B,2,T)
    float* out_sync = out_cert + (size_t)Bb * 2 * Tt;        // (B,NS)

    // ---- workspace bump allocator (256B aligned) ----
    char* wp = (char*)d_ws;
    size_t off = 0;
    auto alloc = [&](size_t bytes) -> void* {
        void* p = wp + off;
        off = (off + bytes + 255) & ~(size_t)255;
        return p;
    };
    const int BS = Bb * Ss;                                  // 12544
    const int OUTp = 1024;                                   // Wout Npad
    float* kv    = (float*)alloc((size_t)BS * Ee * 4);
    float* kh    = (float*)alloc((size_t)BS * Ee * 4);
    float* vh    = (float*)alloc((size_t)BS * Ee * 4);
    float* trace = (float*)alloc((size_t)Bb * Dd * Mm * 4);
    float* act   = (float*)alloc((size_t)Bb * Dd * 4);
    float* aA    = (float*)alloc((size_t)Bb * NSs * 4);
    float* bA    = (float*)alloc((size_t)Bb * NSs * 4);
    float* syncA = (float*)alloc((size_t)Bb * NSs * 4);
    float* aO    = (float*)alloc((size_t)Bb * NSs * 4);
    float* bO    = (float*)alloc((size_t)Bb * NSs * 4);
    float* syncO = (float*)alloc((size_t)Bb * NSs * 4);
    float* rA    = (float*)alloc((size_t)NSs * 4);
    float* rO    = (float*)alloc((size_t)NSs * 4);
    float* qbuf  = (float*)alloc((size_t)Bb * Ee * 4);
    float* qh    = (float*)alloc((size_t)Bb * Ee * 4);
    float* attnb = (float*)alloc((size_t)Bb * Ee * 4);
    float* attnO = (float*)alloc((size_t)Bb * Ee * 4);
    float* pre   = (float*)alloc((size_t)Bb * (Ee + Dd) * 4);
    float* praw  = (float*)alloc((size_t)Bb * 2 * Dd * 4);
    float* predb = (float*)alloc((size_t)Bb * OUTn * 4);
    __bf16* WkvB  = (__bf16*)alloc((size_t)Ee * Ee * 2);
    __bf16* WqB   = (__bf16*)alloc((size_t)NSs * Ee * 2);
    __bf16* WqaB  = (__bf16*)alloc((size_t)Ee * Ee * 2);
    __bf16* WkaB  = (__bf16*)alloc((size_t)Ee * Ee * 2);
    __bf16* WvaB  = (__bf16*)alloc((size_t)Ee * Ee * 2);
    __bf16* WoB   = (__bf16*)alloc((size_t)Ee * Ee * 2);
    __bf16* WsynB = (__bf16*)alloc((size_t)(Ee + Dd) * 2 * Dd * 2);
    __bf16* WoutB = (__bf16*)alloc((size_t)NSs * OUTp * 2);

    // ---- weight swizzle+convert (fp32 -> swizzled bf16, once per call) ----
    auto packw = [&](const float* s, __bf16* dst, int K, int N, int Npad) {
        int n = K * Npad;
        pack_w<<<(n + 255) / 256, 256, 0, stream>>>(s, dst, N, Npad);
    };
    packw(Wkv,  WkvB,  Ee, Ee, Ee);
    packw(Wq,   WqB,   NSs, Ee, Ee);
    packw(Wqa,  WqaB,  Ee, Ee, Ee);
    packw(Wka,  WkaB,  Ee, Ee, Ee);
    packw(Wva,  WvaB,  Ee, Ee, Ee);
    packw(Wo,   WoB,   Ee, Ee, Ee);
    packw(Wsyn, WsynB, Ee + Dd, 2 * Dd, 2 * Dd);
    packw(Wout, WoutB, NSs, OUTn, OUTp);

    // ---- precompute: kv = LN(x@Wkv+bkv); kh = kv@Wka+bka; vh = kv@Wva+bva ----
    gemm_wmma_bf16<<<dim3(Ee / GBN, BS / GBM), 256, 0, stream>>>(
        x, Ee, WkvB, bkv, kv, Ee, Ee, Ee);
    ln_rows512<<<BS, 256, 0, stream>>>(kv, g_kv, b_kv);
    gemm_wmma_bf16<<<dim3(Ee / GBN, BS / GBM), 256, 0, stream>>>(
        kv, Ee, WkaB, bka, kh, Ee, Ee, Ee);
    gemm_wmma_bf16<<<dim3(Ee / GBN, BS / GBM), 256, 0, stream>>>(
        kv, Ee, WvaB, bva, vh, Ee, Ee, Ee);

    // ---- state init ----
    init_bd<<<(Bb * Dd) / 256, 256, 0, stream>>>(start_act, start_trace, act, trace);
    init_ns<<<(Bb * NSs) / 256, 256, 0, stream>>>(start_act, idx_out_l, idx_out_r,
                                                  aA, bA, aO, bO);
    init_decay<<<(NSs + 255) / 256, 256, 0, stream>>>(decay_action, decay_out, rA, rO);

    // ---- sequential thought steps ----
    for (int t = 0; t < Tt; ++t) {
        sync_update<<<(Bb * NSs) / 256, 256, 0, stream>>>(
            act, idx_act_l, idx_act_r, rA, aA, bA, syncA, nullptr);
        gemm_wmma_bf16<<<dim3(Ee / GBN, Bb / GBM), 256, 0, stream>>>(
            syncA, NSs, WqB, bq, qbuf, Ee, Ee, NSs);
        gemm_wmma_bf16<<<dim3(Ee / GBN, Bb / GBM), 256, 0, stream>>>(
            qbuf, Ee, WqaB, bqa, qh, Ee, Ee, Ee);
        attn_kernel<<<Bb * NHh, 256, 0, stream>>>(qh, kh, vh, attnb);
        gemm_wmma_bf16<<<dim3(Ee / GBN, Bb / GBM), 256, 0, stream>>>(
            attnb, Ee, WoB, bo, attnO, Ee, Ee, Ee);
        concat_pre<<<(Bb * (Ee + Dd)) / 256, 256, 0, stream>>>(attnO, act, pre);
        gemm_wmma_bf16<<<dim3((2 * Dd) / GBN, Bb / GBM), 256, 0, stream>>>(
            pre, Ee + Dd, WsynB, bsyn, praw, 2 * Dd, 2 * Dd, Ee + Dd);
        glu_ln_trace<<<Bb, 256, 0, stream>>>(praw, g_syn, b_syn, trace, t % Mm);
        nlm_kernel<<<(Bb * Dd) / 256, 256, 0, stream>>>(
            trace, w1, b1, w2, b2, act, (t + 1) % Mm);
        sync_update<<<(Bb * NSs) / 256, 256, 0, stream>>>(
            act, idx_out_l, idx_out_r, rO, aO, bO, syncO, out_sync);
        gemm_wmma_bf16<<<dim3(OUTp / GBN, Bb / GBM), 256, 0, stream>>>(
            syncO, NSs, WoutB, bout, predb, OUTn, OUTp, NSs);
        pred_final<<<Bb, 256, 0, stream>>>(predb, out_pred, out_cert, t);
    }
}
